// HGAT_4750233829662
// MI455X (gfx1250) — compile-verified, matrix operand order checked by reading, not run on hardware
//
#include <hip/hip_runtime.h>
#include <cstdint>

#define NTYPE_C 3
#define NFEAT_C 300
#define KPAD_C  320
#define NHID_C  512
#define DIM2_C  18
#define ATTH_C  50
#define GAMMA_C 0.1f
#define SLOPE_C 0.2f
#define NEGINF_C (-9.0e15f)

typedef __attribute__((ext_vector_type(16))) __bf16 v16bf;
typedef __attribute__((ext_vector_type(8)))  float  v8f;

union FragBF { v16bf v; unsigned u[8]; };

__device__ __forceinline__ unsigned short f2bf(float f) {
  union { float f; unsigned u; } c; c.f = f;
  unsigned u = c.u;
  u = u + 0x7FFFu + ((u >> 16) & 1u);       // round-to-nearest-even
  return (unsigned short)(u >> 16);
}
__device__ __forceinline__ unsigned pack2(float lo, float hi) {
  return (unsigned)f2bf(lo) | ((unsigned)f2bf(hi) << 16);
}
__device__ __forceinline__ float leakyf(float x) { return x > 0.f ? x : SLOPE_C * x; }

__device__ __forceinline__ v8f wmma_bf16(v16bf a, v16bf b, v8f c) {
  return __builtin_amdgcn_wmma_f32_16x16x32_bf16(false, a, false, b, (short)0, c, false, false);
}

// ---------------------------------------------------------------------------
// 0) one-time: transpose W1[t] (300x512 fp32) -> bf16 [512][320] (K padded)
// ---------------------------------------------------------------------------
__global__ void hgat_w1t(const float* __restrict__ W, unsigned short* __restrict__ WT)
{
  int idx = blockIdx.x * blockDim.x + threadIdx.x;
  if (idx >= NHID_C * KPAD_C) return;
  int n = idx / KPAD_C, k = idx - n * KPAD_C;
  float v = (k < NFEAT_C) ? W[(size_t)k * NHID_C + n] : 0.f;
  WT[idx] = f2bf(v);
}

// ---------------------------------------------------------------------------
// 1) h = x @ W1[t]  (M x 300)@(300 x 512) bf16 WMMA; fp32 H + bf16 HT[n*M+m].
//    128 thr = 4 waves; 16 rows x 64 cols per block; K padded to 320.
// ---------------------------------------------------------------------------
__global__ void __launch_bounds__(128)
hgat_proj_gemm(const float* __restrict__ X, const unsigned short* __restrict__ WT,
               float* __restrict__ H, unsigned short* __restrict__ HT, int M)
{
  __shared__ alignas(16) unsigned Xs[16 * 16];      // 16 rows x 32 bf16
  const int tid = threadIdx.x, lane = tid & 31, wv = tid >> 5;
  const int hh = lane >> 4, mr = lane & 15;
  const int row0 = blockIdx.y * 16;
  const int nc = blockIdx.x * 64 + wv * 16 + mr;

  const int pm = tid >> 3, pk = (tid & 7) * 4;      // 4 contiguous k per thread
  const int prow = row0 + pm;
  const bool rowok = prow < M;
  const float* Xrow = X + (size_t)(rowok ? prow : (M - 1)) * NFEAT_C;

  const unsigned* wtp = (const unsigned*)WT;
  const unsigned boff = (unsigned)nc * (KPAD_C / 2) + (hh ? 4u : 0u);
  const int hhoff = hh ? 4 : 0;

  v8f acc = {0.f,0.f,0.f,0.f,0.f,0.f,0.f,0.f};
  for (int k0 = 0; k0 < KPAD_C; k0 += 32) {
    int ks = k0 + pk;
    int ksc = (ks <= NFEAT_C - 4) ? ks : (NFEAT_C - 4);
    float4 xv = *(const float4*)(Xrow + ksc);
    const bool okk = rowok && (ks < NFEAT_C);
    float x0 = okk ? xv.x : 0.f, x1 = okk ? xv.y : 0.f;
    float x2 = okk ? xv.z : 0.f, x3 = okk ? xv.w : 0.f;
    uint2 pkd; pkd.x = pack2(x0, x1); pkd.y = pack2(x2, x3);
    __syncthreads();
    *(uint2*)&Xs[pm * 16 + (pk >> 1)] = pkd;
    __syncthreads();
    FragBF a, b;
    const uint4* ap = (const uint4*)&Xs[mr * 16 + hhoff];
    uint4 alo = ap[0], ahi = ap[2];
    a.u[0]=alo.x; a.u[1]=alo.y; a.u[2]=alo.z; a.u[3]=alo.w;
    a.u[4]=ahi.x; a.u[5]=ahi.y; a.u[6]=ahi.z; a.u[7]=ahi.w;
    uint4 blo = *(const uint4*)(wtp + boff + (k0 >> 1));
    uint4 bhi = *(const uint4*)(wtp + boff + (k0 >> 1) + 8);
    b.u[0]=blo.x; b.u[1]=blo.y; b.u[2]=blo.z; b.u[3]=blo.w;
    b.u[4]=bhi.x; b.u[5]=bhi.y; b.u[6]=bhi.z; b.u[7]=bhi.w;
    acc = wmma_bf16(a.v, b.v, acc);
  }
#pragma unroll
  for (int r = 0; r < 8; ++r) {
    int gr = row0 + r + 8 * hh;
    if (gr < M) {
      H[(size_t)gr * NHID_C + nc] = acc[r];
      HT[(size_t)nc * M + gr] = f2bf(acc[r]);
    }
  }
}

// ---------------------------------------------------------------------------
// 2) out[r] = dot(H[r,:], a)
// ---------------------------------------------------------------------------
__global__ void hgat_rowdot(const float* __restrict__ H, const float* __restrict__ a,
                            float* __restrict__ out, int M)
{
  int r = blockIdx.x * blockDim.x + threadIdx.x;
  if (r >= M) return;
  const float4* hp = (const float4*)(H + (size_t)r * NHID_C);
  const float4* ap = (const float4*)a;
  float s = 0.f;
  for (int k = 0; k < NHID_C / 4; ++k) {
    float4 h4 = hp[k], a4 = ap[k];
    s += h4.x * a4.x + h4.y * a4.y + h4.z * a4.z + h4.w * a4.w;
  }
  out[r] = s;
}

// ---------------------------------------------------------------------------
// 3) per-row masked softmax stats (max, sum(exp(e-max))) — streams adjacency.
// ---------------------------------------------------------------------------
__global__ void __launch_bounds__(256)
hgat_row_stats(const float* __restrict__ A, const float* __restrict__ f1,
               const float* __restrict__ g, float* __restrict__ rmax,
               float* __restrict__ rsum, int N1, int N2)
{
  __shared__ float red[256];
  const int i = blockIdx.x, tid = threadIdx.x;
  const float fi = f1[i];
  const float* Ar = A + (size_t)i * N2;
  float mx = -3.0e38f;
  for (int j = tid * 4; j < N2; j += 1024) {
    float4 a4 = *(const float4*)(Ar + j);
    float4 g4 = *(const float4*)(g + j);
    float e0 = (a4.x > 0.f) ? leakyf(fi + g4.x) : NEGINF_C;
    float e1 = (a4.y > 0.f) ? leakyf(fi + g4.y) : NEGINF_C;
    float e2 = (a4.z > 0.f) ? leakyf(fi + g4.z) : NEGINF_C;
    float e3 = (a4.w > 0.f) ? leakyf(fi + g4.w) : NEGINF_C;
    mx = fmaxf(mx, fmaxf(fmaxf(e0, e1), fmaxf(e2, e3)));
  }
  red[tid] = mx; __syncthreads();
  for (int s = 128; s > 0; s >>= 1) { if (tid < s) red[tid] = fmaxf(red[tid], red[tid + s]); __syncthreads(); }
  mx = red[0]; __syncthreads();
  float sm = 0.f;
  for (int j = tid * 4; j < N2; j += 1024) {
    float4 a4 = *(const float4*)(Ar + j);
    float4 g4 = *(const float4*)(g + j);
    float e0 = (a4.x > 0.f) ? leakyf(fi + g4.x) : NEGINF_C;
    float e1 = (a4.y > 0.f) ? leakyf(fi + g4.y) : NEGINF_C;
    float e2 = (a4.z > 0.f) ? leakyf(fi + g4.z) : NEGINF_C;
    float e3 = (a4.w > 0.f) ? leakyf(fi + g4.w) : NEGINF_C;
    sm += __expf(e0 - mx) + __expf(e1 - mx) + __expf(e2 - mx) + __expf(e3 - mx);
  }
  red[tid] = sm; __syncthreads();
  for (int s = 128; s > 0; s >>= 1) { if (tid < s) red[tid] += red[tid + s]; __syncthreads(); }
  if (tid == 0) { rmax[i] = mx; rsum[i] = red[0]; }
}

// ---------------------------------------------------------------------------
// 4) fused P-build + WMMA GEMM: out = (g*softmax(mask(e)) + (1-g)*A) @ h2.
//    Branchless, float4 A loads, register-pipelined; guarded tail tile only.
// ---------------------------------------------------------------------------
__global__ void __launch_bounds__(128)
hgat_pv_gemm(const float* __restrict__ A, const float* __restrict__ f1,
             const float* __restrict__ g, const float* __restrict__ rmax,
             const float* __restrict__ rsum, const unsigned short* __restrict__ HT2,
             float* __restrict__ outSlab, int N1, int N2, int rowStride)
{
  __shared__ alignas(16) unsigned Psu[16 * 16];
  const int tid = threadIdx.x, lane = tid & 31, wv = tid >> 5;
  const int hh = lane >> 4, mr = lane & 15;
  const int row0 = blockIdx.y * 16;
  const int nc = blockIdx.x * 64 + wv * 16 + mr;
  const int hhoff = hh ? 4 : 0;

  const int pm = tid >> 3, pk = (tid & 7) * 4;
  const int prow = row0 + pm;
  const bool rowok = prow < N1;
  const int crow = rowok ? prow : (N1 - 1);
  const float fi = f1[crow];
  const float mi = rmax[crow];
  const float inv_si = GAMMA_C / rsum[crow];
  const float* Ab = A + (size_t)crow * N2;

  const unsigned* htp = (const unsigned*)HT2;
  const size_t bcol = (size_t)nc * N2;
  const size_t bdw = (bcol >> 1) + (size_t)hhoff;

  v8f acc = {0.f,0.f,0.f,0.f,0.f,0.f,0.f,0.f};
  const int N2full = N2 & ~31;

  float4 aR{0.f,0.f,0.f,0.f}, gR{0.f,0.f,0.f,0.f};
  if (N2full > 0) { aR = *(const float4*)(Ab + pk); gR = *(const float4*)(g + pk); }

  int k0 = 0;
  for (; k0 < N2full; k0 += 32) {
    // P values from current registers (branchless)
    float e0 = (aR.x > 0.f) ? leakyf(fi + gR.x) : NEGINF_C;
    float e1 = (aR.y > 0.f) ? leakyf(fi + gR.y) : NEGINF_C;
    float e2 = (aR.z > 0.f) ? leakyf(fi + gR.z) : NEGINF_C;
    float e3 = (aR.w > 0.f) ? leakyf(fi + gR.w) : NEGINF_C;
    float p0 = __expf(e0 - mi) * inv_si + (1.f - GAMMA_C) * aR.x;
    float p1 = __expf(e1 - mi) * inv_si + (1.f - GAMMA_C) * aR.y;
    float p2 = __expf(e2 - mi) * inv_si + (1.f - GAMMA_C) * aR.z;
    float p3 = __expf(e3 - mi) * inv_si + (1.f - GAMMA_C) * aR.w;
    if (!rowok) { p0 = p1 = p2 = p3 = 0.f; }
    uint2 pkd; pkd.x = pack2(p0, p1); pkd.y = pack2(p2, p3);
    // pipeline: issue next tile's loads now
    int kn = (k0 + 32 < N2full) ? (k0 + 32) : k0;
    aR = *(const float4*)(Ab + kn + pk);
    gR = *(const float4*)(g + kn + pk);
    __syncthreads();
    *(uint2*)&Psu[pm * 16 + (pk >> 1)] = pkd;
    __syncthreads();
    FragBF a, b;
    const uint4* ap = (const uint4*)&Psu[mr * 16 + hhoff];
    uint4 alo = ap[0], ahi = ap[2];
    a.u[0]=alo.x; a.u[1]=alo.y; a.u[2]=alo.z; a.u[3]=alo.w;
    a.u[4]=ahi.x; a.u[5]=ahi.y; a.u[6]=ahi.z; a.u[7]=ahi.w;
    uint4 blo = *(const uint4*)(htp + bdw + (k0 >> 1));
    uint4 bhi = *(const uint4*)(htp + bdw + (k0 >> 1) + 8);
    b.u[0]=blo.x; b.u[1]=blo.y; b.u[2]=blo.z; b.u[3]=blo.w;
    b.u[4]=bhi.x; b.u[5]=bhi.y; b.u[6]=bhi.z; b.u[7]=bhi.w;
    acc = wmma_bf16(a.v, b.v, acc);
  }

  if (k0 < N2) {                         // tail (N2 % 32 = 16 or 24)
    int gk = k0 + pk;
    int gkc = (gk <= N2 - 4) ? gk : (N2 - 4);
    float4 a4 = *(const float4*)(Ab + gkc);
    float4 g4 = *(const float4*)(g + gkc);
    const bool okk = rowok && (gk < N2);
    float e0 = (a4.x > 0.f) ? leakyf(fi + g4.x) : NEGINF_C;
    float e1 = (a4.y > 0.f) ? leakyf(fi + g4.y) : NEGINF_C;
    float e2 = (a4.z > 0.f) ? leakyf(fi + g4.z) : NEGINF_C;
    float e3 = (a4.w > 0.f) ? leakyf(fi + g4.w) : NEGINF_C;
    float p0 = okk ? (__expf(e0 - mi) * inv_si + (1.f - GAMMA_C) * a4.x) : 0.f;
    float p1 = okk ? (__expf(e1 - mi) * inv_si + (1.f - GAMMA_C) * a4.y) : 0.f;
    float p2 = okk ? (__expf(e2 - mi) * inv_si + (1.f - GAMMA_C) * a4.z) : 0.f;
    float p3 = okk ? (__expf(e3 - mi) * inv_si + (1.f - GAMMA_C) * a4.w) : 0.f;
    uint2 pkd; pkd.x = pack2(p0, p1); pkd.y = pack2(p2, p3);
    __syncthreads();
    *(uint2*)&Psu[pm * 16 + (pk >> 1)] = pkd;
    __syncthreads();
    FragBF a, b;
    const uint4* ap = (const uint4*)&Psu[mr * 16 + hhoff];
    uint4 alo = ap[0], ahi = ap[2];
    a.u[0]=alo.x; a.u[1]=alo.y; a.u[2]=alo.z; a.u[3]=alo.w;
    a.u[4]=ahi.x; a.u[5]=ahi.y; a.u[6]=ahi.z; a.u[7]=ahi.w;
#pragma unroll
    for (int v = 0; v < 4; ++v) {
      int kb = hhoff * 2 + 2 * v;
      int kk0 = k0 + kb, kk1 = k0 + 16 + kb;
      int kk0c = (kk0 < N2) ? kk0 : (N2 - 2);
      int kk1c = (kk1 < N2) ? kk1 : (N2 - 2);
      unsigned b0 = htp[(bcol + (size_t)kk0c) >> 1];
      unsigned b1 = htp[(bcol + (size_t)kk1c) >> 1];
      b.u[v]     = (kk0 < N2) ? b0 : 0u;
      b.u[v + 4] = (kk1 < N2) ? b1 : 0u;
    }
    acc = wmma_bf16(a.v, b.v, acc);
  }

#pragma unroll
  for (int r = 0; r < 8; ++r) {
    int gr = row0 + r + 8 * hh;
    if (gr < N1) outSlab[(size_t)gr * rowStride + nc] = acc[r];
  }
}

// ---------------------------------------------------------------------------
// 5) type-level self-attention layer 1 + relu; one wave per node.
// ---------------------------------------------------------------------------
__global__ void __launch_bounds__(128)
hgat_self_att1(const float* __restrict__ stk, const float* __restrict__ Wat,
               const float* __restrict__ bat, const float* __restrict__ vat,
               float* __restrict__ lay, int Nn, int tSelf)
{
  __shared__ float Z[4][NTYPE_C * ATTH_C];
  __shared__ float SC[4][NTYPE_C];
  const int lane = threadIdx.x & 31, wv = threadIdx.x >> 5;
  const int onode = blockIdx.x * 4 + wv;
  const bool live = onode < Nn;
  const int node = live ? onode : (Nn - 1);
  const float* srow = stk + (size_t)node * (NTYPE_C * NHID_C);

  for (int idx = lane; idx < NTYPE_C * ATTH_C; idx += 32) {
    int t = idx / ATTH_C, k = idx % ATTH_C;
    float acc = bat[k];
    const float* sp = srow + t * NHID_C;
    for (int f = 0; f < NHID_C; f += 4) {
      float4 s4 = *(const float4*)(sp + f);
      acc += s4.x * Wat[f * ATTH_C + k] + s4.y * Wat[(f + 1) * ATTH_C + k]
           + s4.z * Wat[(f + 2) * ATTH_C + k] + s4.w * Wat[(f + 3) * ATTH_C + k];
    }
    Z[wv][t * ATTH_C + k] = tanhf(acc);
  }
  __syncthreads();
  if (lane < NTYPE_C) {
    float sc = 0.f;
    for (int k = 0; k < ATTH_C; ++k)
      sc += Z[wv][tSelf * ATTH_C + k] * vat[k] + Z[wv][lane * ATTH_C + k] * vat[ATTH_C + k];
    SC[wv][lane] = leakyf(sc);
  }
  __syncthreads();
  float m = fmaxf(SC[wv][0], fmaxf(SC[wv][1], SC[wv][2]));
  float e0 = __expf(SC[wv][0] - m), e1 = __expf(SC[wv][1] - m), e2 = __expf(SC[wv][2] - m);
  float inv = 1.f / (e0 + e1 + e2);
  float a0 = e0 * inv, a1 = e1 * inv, a2 = e2 * inv;
  for (int f = lane; f < NHID_C; f += 32) {
    float o = a0 * srow[f] + a1 * srow[NHID_C + f] + a2 * srow[2 * NHID_C + f];
    if (live) lay[(size_t)onode * NHID_C + f] = fmaxf(o, 0.f);
  }
}

// ---------------------------------------------------------------------------
// 6) sup = lay1 @ W2 (512->18) -> bf16 transposed, padded to 32 rows.
// ---------------------------------------------------------------------------
__global__ void hgat_sup(const float* __restrict__ lay, const float* __restrict__ W2,
                         unsigned short* __restrict__ supT, int Nn)
{
  int idx = blockIdx.x * blockDim.x + threadIdx.x;
  if (idx >= 32 * Nn) return;
  int d = idx / Nn, row = idx - d * Nn;
  float v = 0.f;
  if (d < DIM2_C) {
    const float4* lp = (const float4*)(lay + (size_t)row * NHID_C);
    for (int f = 0; f < NHID_C / 4; ++f) {
      float4 l4 = lp[f];
      v += l4.x * W2[(4 * f) * DIM2_C + d]     + l4.y * W2[(4 * f + 1) * DIM2_C + d]
         + l4.z * W2[(4 * f + 2) * DIM2_C + d] + l4.w * W2[(4 * f + 3) * DIM2_C + d];
    }
  }
  supT[(size_t)d * Nn + row] = f2bf(v);
}

// ---------------------------------------------------------------------------
// 7) stk2 slab = A @ sup + b2; WMMA bf16; 64 thr = 2 waves; N padded 18->32.
// ---------------------------------------------------------------------------
__global__ void __launch_bounds__(64)
hgat_adj_sup(const float* __restrict__ A, const unsigned short* __restrict__ supT,
             const float* __restrict__ b2, float* __restrict__ outSlab,
             int N1, int N2, int rowStride)
{
  __shared__ alignas(16) unsigned Asu[16 * 16];
  const int tid = threadIdx.x, lane = tid & 31, wv = tid >> 5;
  const int hh = lane >> 4, mr = lane & 15;
  const int row0 = blockIdx.x * 16;
  const int nc = wv * 16 + mr;            // 0..31
  const int hhoff = hh ? 4 : 0;

  const int pm = tid >> 2, pk = (tid & 3) * 8;   // 8 contiguous k per thread
  const int prow = row0 + pm;
  const bool rowok = prow < N1;
  const float* Ab = A + (size_t)(rowok ? prow : (N1 - 1)) * N2;

  const unsigned* sp = (const unsigned*)supT;
  const size_t bcol = (size_t)nc * N2;
  const size_t bdw = (bcol >> 1) + (size_t)hhoff;

  v8f acc = {0.f,0.f,0.f,0.f,0.f,0.f,0.f,0.f};
  const int N2full = N2 & ~31;

  float4 aR0{0.f,0.f,0.f,0.f}, aR1{0.f,0.f,0.f,0.f};
  if (N2full > 0) { aR0 = *(const float4*)(Ab + pk); aR1 = *(const float4*)(Ab + pk + 4); }

  int k0 = 0;
  for (; k0 < N2full; k0 += 32) {
    float v0 = rowok ? aR0.x : 0.f, v1 = rowok ? aR0.y : 0.f;
    float v2 = rowok ? aR0.z : 0.f, v3 = rowok ? aR0.w : 0.f;
    float v4 = rowok ? aR1.x : 0.f, v5 = rowok ? aR1.y : 0.f;
    float v6 = rowok ? aR1.z : 0.f, v7 = rowok ? aR1.w : 0.f;
    uint4 pkd;
    pkd.x = pack2(v0, v1); pkd.y = pack2(v2, v3);
    pkd.z = pack2(v4, v5); pkd.w = pack2(v6, v7);
    int kn = (k0 + 32 < N2full) ? (k0 + 32) : k0;
    aR0 = *(const float4*)(Ab + kn + pk);
    aR1 = *(const float4*)(Ab + kn + pk + 4);
    __syncthreads();
    *(uint4*)&Asu[pm * 16 + (pk >> 1)] = pkd;
    __syncthreads();
    FragBF a, b;
    const uint4* ap = (const uint4*)&Asu[mr * 16 + hhoff];
    uint4 alo = ap[0], ahi = ap[2];
    a.u[0]=alo.x; a.u[1]=alo.y; a.u[2]=alo.z; a.u[3]=alo.w;
    a.u[4]=ahi.x; a.u[5]=ahi.y; a.u[6]=ahi.z; a.u[7]=ahi.w;
    uint4 blo = *(const uint4*)(sp + bdw + (k0 >> 1));
    uint4 bhi = *(const uint4*)(sp + bdw + (k0 >> 1) + 8);
    b.u[0]=blo.x; b.u[1]=blo.y; b.u[2]=blo.z; b.u[3]=blo.w;
    b.u[4]=bhi.x; b.u[5]=bhi.y; b.u[6]=bhi.z; b.u[7]=bhi.w;
    acc = wmma_bf16(a.v, b.v, acc);
  }

  if (k0 < N2) {                          // tail tile
    int gk = k0 + pk;
    int gkc = (gk <= N2 - 8) ? gk : (N2 - 8);
    float4 a40 = *(const float4*)(Ab + gkc);
    float4 a41 = *(const float4*)(Ab + gkc + 4);
    const bool okk = rowok && (gk < N2);  // N2 % 8 == 0 for all pairs
    float v0 = okk ? a40.x : 0.f, v1 = okk ? a40.y : 0.f;
    float v2 = okk ? a40.z : 0.f, v3 = okk ? a40.w : 0.f;
    float v4 = okk ? a41.x : 0.f, v5 = okk ? a41.y : 0.f;
    float v6 = okk ? a41.z : 0.f, v7 = okk ? a41.w : 0.f;
    uint4 pkd;
    pkd.x = pack2(v0, v1); pkd.y = pack2(v2, v3);
    pkd.z = pack2(v4, v5); pkd.w = pack2(v6, v7);
    __syncthreads();
    *(uint4*)&Asu[pm * 16 + (pk >> 1)] = pkd;
    __syncthreads();
    FragBF a, b;
    const uint4* ap = (const uint4*)&Asu[mr * 16 + hhoff];
    uint4 alo = ap[0], ahi = ap[2];
    a.u[0]=alo.x; a.u[1]=alo.y; a.u[2]=alo.z; a.u[3]=alo.w;
    a.u[4]=ahi.x; a.u[5]=ahi.y; a.u[6]=ahi.z; a.u[7]=ahi.w;
#pragma unroll
    for (int v = 0; v < 4; ++v) {
      int kb = hhoff * 2 + 2 * v;
      int kk0 = k0 + kb, kk1 = k0 + 16 + kb;
      int kk0c = (kk0 < N2) ? kk0 : (N2 - 2);
      int kk1c = (kk1 < N2) ? kk1 : (N2 - 2);
      unsigned bb0 = sp[(bcol + (size_t)kk0c) >> 1];
      unsigned bb1 = sp[(bcol + (size_t)kk1c) >> 1];
      b.u[v]     = (kk0 < N2) ? bb0 : 0u;
      b.u[v + 4] = (kk1 < N2) ? bb1 : 0u;
    }
    acc = wmma_bf16(a.v, b.v, acc);
  }

  if (nc < DIM2_C) {
    float bias = b2[nc];
#pragma unroll
    for (int r = 0; r < 8; ++r) {
      int gr = row0 + r + 8 * hh;
      if (gr < N1) outSlab[(size_t)gr * rowStride + nc] = acc[r] + bias;
    }
  }
}

// ---------------------------------------------------------------------------
// 8) type attention layer 2 + log_softmax -> output; thread per node.
// ---------------------------------------------------------------------------
__global__ void hgat_self_att2(const float* __restrict__ stk2, const float* __restrict__ Wat,
                               const float* __restrict__ bat, const float* __restrict__ vat,
                               float* __restrict__ out, int Nn, int tSelf)
{
  int n = blockIdx.x * blockDim.x + threadIdx.x;
  if (n >= Nn) return;
  const float* s = stk2 + (size_t)n * (NTYPE_C * DIM2_C);
  float z[NTYPE_C][ATTH_C];
  for (int t = 0; t < NTYPE_C; ++t)
    for (int k = 0; k < ATTH_C; ++k) {
      float acc = bat[k];
      for (int d = 0; d < DIM2_C; ++d) acc += s[t * DIM2_C + d] * Wat[d * ATTH_C + k];
      z[t][k] = tanhf(acc);
    }
  float sc[NTYPE_C];
  for (int t = 0; t < NTYPE_C; ++t) {
    float v = 0.f;
    for (int k = 0; k < ATTH_C; ++k) v += z[tSelf][k] * vat[k] + z[t][k] * vat[ATTH_C + k];
    sc[t] = leakyf(v);
  }
  float m = fmaxf(sc[0], fmaxf(sc[1], sc[2]));
  float e0 = __expf(sc[0] - m), e1 = __expf(sc[1] - m), e2 = __expf(sc[2] - m);
  float inv = 1.f / (e0 + e1 + e2);
  float a0 = e0 * inv, a1 = e1 * inv, a2 = e2 * inv;
  float o[DIM2_C];
  float om = -3.0e38f;
  for (int d = 0; d < DIM2_C; ++d) {
    o[d] = a0 * s[d] + a1 * s[DIM2_C + d] + a2 * s[2 * DIM2_C + d];
    om = fmaxf(om, o[d]);
  }
  float se = 0.f;
  for (int d = 0; d < DIM2_C; ++d) se += __expf(o[d] - om);
  float ls = __logf(se);
  for (int d = 0; d < DIM2_C; ++d) out[(size_t)n * DIM2_C + d] = o[d] - om - ls;
}

// ---------------------------------------------------------------------------
extern "C" void kernel_launch(void* const* d_in, const int* in_sizes, int n_in,
                              void* d_out, int out_size, void* d_ws, size_t ws_size,
                              hipStream_t stream)
{
  (void)in_sizes; (void)n_in; (void)out_size; (void)ws_size;
  const int NSl[3] = {4000, 3000, 2000};
  const int PRE[3] = {0, 4000, 7000};
  const int NTOT   = 9000;

  const float* x[3];
  for (int t = 0; t < 3; ++t) x[t] = (const float*)d_in[t];
  const float* adj[3][3];
  for (int t1 = 0; t1 < 3; ++t1)
    for (int t2 = 0; t2 < 3; ++t2) adj[t1][t2] = (const float*)d_in[3 + t1 * 3 + t2];
  const float* W1   = (const float*)d_in[12];
  const float* a1   = (const float*)d_in[13];
  const float* a2   = (const float*)d_in[14];
  const float* Wat1 = (const float*)d_in[15];
  const float* bat1 = (const float*)d_in[16];
  const float* vat1 = (const float*)d_in[17];
  const float* W2   = (const float*)d_in[18];
  const float* b2   = (const float*)d_in[19];
  const float* Wat2 = (const float*)d_in[20];
  const float* bat2 = (const float*)d_in[21];
  const float* vat2 = (const float*)d_in[22];

  char* p = (char*)d_ws;
  auto carve = [&](size_t bytes) -> char* {
    char* r = p; p += (bytes + 255) & ~(size_t)255; return r;
  };
  float* h[3]; unsigned short* hT[3]; unsigned short* w1T[3];
  for (int t = 0; t < 3; ++t) h[t]   = (float*)carve((size_t)NSl[t] * NHID_C * 4);
  for (int t = 0; t < 3; ++t) hT[t]  = (unsigned short*)carve((size_t)NSl[t] * NHID_C * 2);
  for (int t = 0; t < 3; ++t) w1T[t] = (unsigned short*)carve((size_t)NHID_C * KPAD_C * 2);
  float* f1   = (float*)carve((size_t)NTOT * 4);
  float* gv   = (float*)carve((size_t)3 * NTOT * 4);
  float* rmax = (float*)carve((size_t)3 * NTOT * 4);
  float* rsum = (float*)carve((size_t)3 * NTOT * 4);
  float* stk1 = (float*)carve((size_t)NTOT * 3 * NHID_C * 4);
  float* lay1 = (float*)carve((size_t)NTOT * NHID_C * 4);
  unsigned short* supT = (unsigned short*)carve((size_t)32 * NTOT * 2);
  float* stk2 = (float*)carve((size_t)NTOT * 3 * DIM2_C * 4);

  // 0) W1 -> bf16 transposed, K padded
  for (int t = 0; t < 3; ++t)
    hgat_w1t<<<(NHID_C * KPAD_C + 255) / 256, 256, 0, stream>>>(
        W1 + (size_t)t * NFEAT_C * NHID_C, w1T[t]);
  // 1) per-type projections
  for (int t = 0; t < 3; ++t) {
    dim3 grid(NHID_C / 64, (NSl[t] + 15) / 16);
    hgat_proj_gemm<<<grid, 128, 0, stream>>>(x[t], w1T[t], h[t], hT[t], NSl[t]);
  }
  // 2) attention dot-product vectors
  for (int t1 = 0; t1 < 3; ++t1)
    hgat_rowdot<<<(NSl[t1] + 255) / 256, 256, 0, stream>>>(h[t1], a1 + t1 * NHID_C,
                                                           f1 + PRE[t1], NSl[t1]);
  for (int t1 = 0; t1 < 3; ++t1)
    for (int t2 = 0; t2 < 3; ++t2)
      hgat_rowdot<<<(NSl[t2] + 255) / 256, 256, 0, stream>>>(h[t2], a2 + t1 * NHID_C,
                                                             gv + t1 * NTOT + PRE[t2], NSl[t2]);
  // 3) masked softmax row stats
  for (int t1 = 0; t1 < 3; ++t1)
    for (int t2 = 0; t2 < 3; ++t2)
      hgat_row_stats<<<NSl[t1], 256, 0, stream>>>(adj[t1][t2], f1 + PRE[t1],
                                                  gv + t1 * NTOT + PRE[t2],
                                                  rmax + 3 * PRE[t1] + t2 * NSl[t1],
                                                  rsum + 3 * PRE[t1] + t2 * NSl[t1],
                                                  NSl[t1], NSl[t2]);
  // 4) fused P @ h (dominant WMMA phase)
  for (int t1 = 0; t1 < 3; ++t1)
    for (int t2 = 0; t2 < 3; ++t2) {
      dim3 grid(NHID_C / 64, (NSl[t1] + 15) / 16);
      hgat_pv_gemm<<<grid, 128, 0, stream>>>(adj[t1][t2], f1 + PRE[t1],
                                             gv + t1 * NTOT + PRE[t2],
                                             rmax + 3 * PRE[t1] + t2 * NSl[t1],
                                             rsum + 3 * PRE[t1] + t2 * NSl[t1],
                                             hT[t2],
                                             stk1 + (size_t)PRE[t1] * 3 * NHID_C + t2 * NHID_C,
                                             NSl[t1], NSl[t2], 3 * NHID_C);
    }
  // 5) type attention layer 1 + relu
  for (int t1 = 0; t1 < 3; ++t1)
    hgat_self_att1<<<(NSl[t1] + 3) / 4, 128, 0, stream>>>(
        stk1 + (size_t)PRE[t1] * 3 * NHID_C,
        Wat1 + (size_t)t1 * NHID_C * ATTH_C, bat1 + t1 * ATTH_C, vat1 + t1 * 2 * ATTH_C,
        lay1 + (size_t)PRE[t1] * NHID_C, NSl[t1], t1);
  // 6) sup = lay1 @ W2 -> bf16 transposed
  for (int t = 0; t < 3; ++t)
    hgat_sup<<<(32 * NSl[t] + 255) / 256, 256, 0, stream>>>(lay1 + (size_t)PRE[t] * NHID_C, W2,
                                                            supT + (size_t)PRE[t] * 32, NSl[t]);
  // 7) stk2 = adj @ sup + b2
  for (int t1 = 0; t1 < 3; ++t1)
    for (int t2 = 0; t2 < 3; ++t2)
      hgat_adj_sup<<<(NSl[t1] + 15) / 16, 64, 0, stream>>>(
          adj[t1][t2], supT + (size_t)PRE[t2] * 32, b2,
          stk2 + (size_t)PRE[t1] * 3 * DIM2_C + t2 * DIM2_C,
          NSl[t1], NSl[t2], 3 * DIM2_C);
  // 8) type attention layer 2 + log_softmax -> d_out
  float* out = (float*)d_out;
  for (int t1 = 0; t1 < 3; ++t1)
    hgat_self_att2<<<(NSl[t1] + 255) / 256, 256, 0, stream>>>(
        stk2 + (size_t)PRE[t1] * 3 * DIM2_C,
        Wat2 + (size_t)t1 * DIM2_C * ATTH_C, bat2 + t1 * ATTH_C, vat2 + t1 * 2 * ATTH_C,
        out + (size_t)PRE[t1] * DIM2_C, NSl[t1], t1);
}
